// transformer_26362509263331
// MI455X (gfx1250) — matrix operand order repaired and checked by reading, not compiled, not run on hardware
//
#include <hip/hip_runtime.h>
#include <hip/hip_bf16.h>

#define D_MODEL  1024
#define D_HIDDEN 4096
#define N_CTX    2048
#define N_LAYERS 8
#define D_VOCAB  32000

typedef __bf16 bf16_t;
typedef bf16_t v16bf __attribute__((ext_vector_type(16)));
typedef bf16_t v8bf  __attribute__((ext_vector_type(8)));
typedef float  v8f   __attribute__((ext_vector_type(8)));

// ---------------------------------------------------------------------------
// Embedding: X[i,d] = tok_emb[tokens[i], d] + pos_emb[i, d]   (f32)
// ---------------------------------------------------------------------------
__global__ void embed_kernel(const int* __restrict__ tokens,
                             const float* __restrict__ tok_emb,
                             const float* __restrict__ pos_emb,
                             float* __restrict__ X)
{
    int idx = blockIdx.x * blockDim.x + threadIdx.x;
    int row = idx / D_MODEL;
    int col = idx - row * D_MODEL;
    if (row < N_CTX)
        X[idx] = tok_emb[(size_t)tokens[row] * D_MODEL + col] + pos_emb[idx];
}

// ---------------------------------------------------------------------------
// Elementwise f32 -> bf16 convert, 4 elements per thread (n % 1024 == 0)
// ---------------------------------------------------------------------------
__global__ void convert_bf16_kernel(const float* __restrict__ in,
                                    bf16_t* __restrict__ out, size_t n)
{
    size_t i = ((size_t)blockIdx.x * blockDim.x + threadIdx.x) * 4;
    if (i + 3 < n) {
        float4 v = *reinterpret_cast<const float4*>(in + i);
        out[i + 0] = (bf16_t)v.x;
        out[i + 1] = (bf16_t)v.y;
        out[i + 2] = (bf16_t)v.z;
        out[i + 3] = (bf16_t)v.w;
    }
}

// ---------------------------------------------------------------------------
// Tiled transpose-convert: in f32 [R,C]  ->  out bf16 [C,R]
// R, C multiples of 32. Block (32,8), grid (C/32, R/32).
// ---------------------------------------------------------------------------
__global__ __launch_bounds__(256)
void transpose_convert_kernel(const float* __restrict__ in,
                              bf16_t* __restrict__ out, int R, int C)
{
    __shared__ bf16_t tile[32][33];
    const int c0 = blockIdx.x * 32;
    const int r0 = blockIdx.y * 32;
    const int tx = threadIdx.x, ty = threadIdx.y;
#pragma unroll
    for (int i = 0; i < 32; i += 8)
        tile[ty + i][tx] = (bf16_t)in[(size_t)(r0 + ty + i) * C + (c0 + tx)];
    __syncthreads();
#pragma unroll
    for (int i = 0; i < 32; i += 8)
        out[(size_t)(c0 + ty + i) * R + (r0 + tx)] = tile[tx][ty + i];
}

// ---------------------------------------------------------------------------
// bf16 WMMA GEMM:  C[M,N] = epi(A[M,K] @ B^T (+ bias))
// A bf16 [M,K] (lda), B bf16 stored [N,K] (ldb) so every lane's fragment
// load is contiguous. Each wave computes a 32x64 block:
//   MT=2 M-tiles x NT=4 N-tiles, f32 accumulation (8 wmma / K-step).
//   EPI: 0 = none, 1 = bias + relu, 2 = bias
//   OBF: true -> C is bf16, else f32
// ---------------------------------------------------------------------------
template<int EPI, bool OBF>
__global__ __launch_bounds__(256)
void gemm_bf16t(const bf16_t* __restrict__ A, const bf16_t* __restrict__ B,
                const float* __restrict__ bias, void* __restrict__ Cv,
                int M, int N, int K, int lda, int ldb, int ldc)
{
    constexpr int MT = 2, NT = 4;
    const int lane = threadIdx.x & 31;
    const int wid  = threadIdx.x >> 5;
    const int l15  = lane & 15;
    const int hi   = lane >> 4;                // 0 or 1

    const int strips_n = (N >> 4) / NT;
    const int strips_m = (M >> 4) / MT;

    int strip = blockIdx.x * 8 + wid;          // wave-uniform
    if (strip >= strips_m * strips_n) return;
    const int sm  = strip / strips_n;
    const int tm0 = sm * MT;
    const int tn0 = (strip - sm * strips_n) * NT;

    v8f acc[MT][NT] = {};

    // A fragment: lane row M = tile*16 + l15; lane-hi picks K 0-15 vs 16-31
    // (index directly off A so the compiler keeps global addressing)
    const size_t aoff0 = (size_t)(tm0 * 16 + l15) * lda + hi * 16;
    const size_t arowstride = (size_t)16 * lda;

    // B fragment: lane col N = tile*16 + l15; 16 contiguous K per lane (hi*16 offset)
    const size_t boff0 = (size_t)(tn0 * 16 + l15) * ldb + hi * 16;
    const size_t btile = (size_t)16 * ldb;

    for (int k0 = 0; k0 < K; k0 += 32) {
        v16bf a[MT];
#pragma unroll
        for (int m = 0; m < MT; ++m) {
            const bf16_t* ap = A + aoff0 + (size_t)m * arowstride + k0;
            v8bf lo = *reinterpret_cast<const v8bf*>(ap);
            v8bf h8 = *reinterpret_cast<const v8bf*>(ap + 8);
#pragma unroll
            for (int i = 0; i < 8; ++i) { a[m][i] = lo[i]; a[m][i + 8] = h8[i]; }
        }
        __builtin_prefetch(B + boff0 + k0 + 96, 0, 1);  // global_prefetch_b8
#pragma unroll
        for (int t = 0; t < NT; ++t) {
            v16bf b = *reinterpret_cast<const v16bf*>(B + boff0 + (size_t)t * btile + k0);
#pragma unroll
            for (int m = 0; m < MT; ++m)
                acc[m][t] = __builtin_amdgcn_wmma_f32_16x16x32_bf16(
                    false, a[m], false, b, (short)0, acc[m][t], false, false);
        }
    }

    // Epilogue: D layout — lane col = l15, VGPR r -> row r + 8*hi
#pragma unroll
    for (int m = 0; m < MT; ++m) {
        const int row0 = (tm0 + m) * 16 + hi * 8;
#pragma unroll
        for (int t = 0; t < NT; ++t) {
            const int col = (tn0 + t) * 16 + l15;
            float bv = 0.0f;
            if (EPI != 0) bv = bias[col];
#pragma unroll
            for (int r = 0; r < 8; ++r) {
                float v = acc[m][t][r] + bv;
                if (EPI == 1) v = fmaxf(v, 0.0f);
                if (OBF) ((bf16_t*)Cv)[(size_t)(row0 + r) * ldc + col] = (bf16_t)v;
                else     ((float*) Cv)[(size_t)(row0 + r) * ldc + col] = v;
            }
        }
    }
}

// ---------------------------------------------------------------------------
// Causal softmax: reads f32 scores S (mutated as scratch), writes bf16 probs P.
// One block (256 threads = 8 wave32) per row; j > row written as 0.
// ---------------------------------------------------------------------------
__global__ __launch_bounds__(256)
void softmax_causal_kernel(float* __restrict__ S, bf16_t* __restrict__ P, int n)
{
    const int row = blockIdx.x;
    float* srow = S + (size_t)row * n;
    bf16_t* prow = P + (size_t)row * n;
    const int tid = threadIdx.x;
    __shared__ float red[8];

    float mx = -3.4e38f;
    for (int j = tid; j <= row; j += 256) mx = fmaxf(mx, srow[j]);
#pragma unroll
    for (int off = 16; off > 0; off >>= 1) mx = fmaxf(mx, __shfl_xor(mx, off, 32));
    if ((tid & 31) == 0) red[tid >> 5] = mx;
    __syncthreads();
    mx = red[0];
#pragma unroll
    for (int w = 1; w < 8; ++w) mx = fmaxf(mx, red[w]);
    __syncthreads();

    float sum = 0.0f;
    for (int j = tid; j <= row; j += 256) {
        float e = __expf(srow[j] - mx);
        srow[j] = e;
        sum += e;
    }
#pragma unroll
    for (int off = 16; off > 0; off >>= 1) sum += __shfl_xor(sum, off, 32);
    if ((tid & 31) == 0) red[tid >> 5] = sum;
    __syncthreads();
    sum = 0.0f;
#pragma unroll
    for (int w = 0; w < 8; ++w) sum += red[w];
    const float inv = 1.0f / sum;

    for (int j = tid; j <= row; j += 256) prow[j] = (bf16_t)(srow[j] * inv);
    for (int j = row + 1 + tid; j < n; j += 256) prow[j] = (bf16_t)0.0f;
}

// ---------------------------------------------------------------------------
// Parallel residual: X += MLP + ATT   (f32)
// ---------------------------------------------------------------------------
__global__ void add3_kernel(float* __restrict__ X, const float* __restrict__ Aa,
                            const float* __restrict__ Bb, int n)
{
    int i = blockIdx.x * blockDim.x + threadIdx.x;
    if (i < n) X[i] = X[i] + Aa[i] + Bb[i];
}

// ---------------------------------------------------------------------------
static inline int gemm_blocks(int M, int N)
{
    int strips = ((M >> 4) / 2) * ((N >> 4) / 4);
    return (strips + 7) / 8;
}

extern "C" void kernel_launch(void* const* d_in, const int* in_sizes, int n_in,
                              void* d_out, int out_size, void* d_ws, size_t ws_size,
                              hipStream_t stream)
{
    (void)in_sizes; (void)n_in; (void)out_size; (void)ws_size;

    const int*   tokens  = (const int*)  d_in[0];
    const float* tok_emb = (const float*)d_in[1];
    const float* pos_emb = (const float*)d_in[2];
    const float* wqk     = (const float*)d_in[3];
    const float* wov     = (const float*)d_in[4];
    const float* w_up    = (const float*)d_in[5];
    const float* b_up    = (const float*)d_in[6];
    const float* w_down  = (const float*)d_in[7];
    const float* b_down  = (const float*)d_in[8];
    float* logits = (float*)d_out;

    // ---- workspace layout ----
    char* p = (char*)d_ws;
    auto alloc_f32 = [&p](size_t n) { float* q = (float*)p; p += n * sizeof(float);  return q; };
    auto alloc_bf  = [&p](size_t n) { bf16_t* q = (bf16_t*)p; p += n * sizeof(bf16_t); return q; };

    float*  X    = alloc_f32((size_t)N_CTX * D_MODEL);    // residual stream (f32)
    float*  S    = alloc_f32((size_t)N_CTX * N_CTX);      // attention scores (f32)
    float*  MLP  = alloc_f32((size_t)N_CTX * D_MODEL);
    float*  ATT  = alloc_f32((size_t)N_CTX * D_MODEL);
    bf16_t* TEb  = alloc_bf((size_t)D_VOCAB * D_MODEL);   // tok_emb bf16 [V,K]
    bf16_t* Xb   = alloc_bf((size_t)N_CTX * D_MODEL);     // X bf16
    bf16_t* Xt   = alloc_bf((size_t)D_MODEL * N_CTX);     // X^T bf16 [D,T]
    bf16_t* H1b  = alloc_bf((size_t)N_CTX * D_HIDDEN);
    bf16_t* Qb   = alloc_bf((size_t)N_CTX * D_MODEL);
    bf16_t* Sb   = alloc_bf((size_t)N_CTX * N_CTX);       // probabilities bf16
    bf16_t* AVb  = alloc_bf((size_t)N_CTX * D_MODEL);
    bf16_t* WUPb = alloc_bf((size_t)D_HIDDEN * D_MODEL);  // per-layer staging
    bf16_t* WDNb = alloc_bf((size_t)D_MODEL * D_HIDDEN);
    bf16_t* WQKt = alloc_bf((size_t)D_MODEL * D_MODEL);   // wqk^T [N,K]
    bf16_t* WOVt = alloc_bf((size_t)D_MODEL * D_MODEL);   // wov^T [N,K]

    const size_t nXd = (size_t)N_CTX * D_MODEL;

    embed_kernel<<<(int)(nXd / 256), 256, 0, stream>>>(tokens, tok_emb, pos_emb, X);

    // tok_emb -> bf16 once (already [N,K] layout for B^T GEMM)
    convert_bf16_kernel<<<(int)(((size_t)D_VOCAB * D_MODEL) / 1024), 256, 0, stream>>>(
        tok_emb, TEb, (size_t)D_VOCAB * D_MODEL);

    for (int l = 0; l < N_LAYERS; ++l) {
        const float* Wup = w_up   + (size_t)l * D_HIDDEN * D_MODEL;
        const float* Bup = b_up   + (size_t)l * D_HIDDEN;
        const float* Wdn = w_down + (size_t)l * D_MODEL * D_HIDDEN;
        const float* Bdn = b_down + (size_t)l * D_MODEL;
        const float* Wqk = wqk    + (size_t)l * D_MODEL * D_MODEL;
        const float* Wov = wov    + (size_t)l * D_MODEL * D_MODEL;

        // stage activations + weights in bf16
        convert_bf16_kernel<<<(int)(nXd / 1024), 256, 0, stream>>>(X, Xb, nXd);
        transpose_convert_kernel<<<dim3(D_MODEL / 32, N_CTX / 32), dim3(32, 8), 0, stream>>>(
            X, Xt, N_CTX, D_MODEL);
        convert_bf16_kernel<<<(int)(((size_t)D_HIDDEN * D_MODEL) / 1024), 256, 0, stream>>>(
            Wup, WUPb, (size_t)D_HIDDEN * D_MODEL);
        convert_bf16_kernel<<<(int)(((size_t)D_MODEL * D_HIDDEN) / 1024), 256, 0, stream>>>(
            Wdn, WDNb, (size_t)D_MODEL * D_HIDDEN);
        transpose_convert_kernel<<<dim3(D_MODEL / 32, D_MODEL / 32), dim3(32, 8), 0, stream>>>(
            Wqk, WQKt, D_MODEL, D_MODEL);
        transpose_convert_kernel<<<dim3(D_MODEL / 32, D_MODEL / 32), dim3(32, 8), 0, stream>>>(
            Wov, WOVt, D_MODEL, D_MODEL);

        // H1 = relu(X @ Wup^T + b_up)          -> bf16
        gemm_bf16t<1, true><<<gemm_blocks(N_CTX, D_HIDDEN), 256, 0, stream>>>(
            Xb, WUPb, Bup, H1b, N_CTX, D_HIDDEN, D_MODEL, D_MODEL, D_MODEL, D_HIDDEN);
        // MLP = H1 @ Wdn^T + b_down            -> f32
        gemm_bf16t<2, false><<<gemm_blocks(N_CTX, D_MODEL), 256, 0, stream>>>(
            H1b, WDNb, Bdn, MLP, N_CTX, D_MODEL, D_HIDDEN, D_HIDDEN, D_HIDDEN, D_MODEL);
        // Q = X @ Wqk  (via Wqk^T staged [N,K]) -> bf16
        gemm_bf16t<0, true><<<gemm_blocks(N_CTX, D_MODEL), 256, 0, stream>>>(
            Xb, WQKt, nullptr, Qb, N_CTX, D_MODEL, D_MODEL, D_MODEL, D_MODEL, D_MODEL);
        // S = Q @ X^T  (B storage = Xb [T,K])   -> f32
        gemm_bf16t<0, false><<<gemm_blocks(N_CTX, N_CTX), 256, 0, stream>>>(
            Qb, Xb, nullptr, S, N_CTX, N_CTX, D_MODEL, D_MODEL, D_MODEL, N_CTX);
        // P = causal_softmax(S)                 -> bf16
        softmax_causal_kernel<<<N_CTX, 256, 0, stream>>>(S, Sb, N_CTX);
        // AV = P @ X   (B storage = Xt [D,T])   -> bf16
        gemm_bf16t<0, true><<<gemm_blocks(N_CTX, D_MODEL), 256, 0, stream>>>(
            Sb, Xt, nullptr, AVb, N_CTX, D_MODEL, N_CTX, N_CTX, N_CTX, D_MODEL);
        // ATT = AV @ Wov (via Wov^T staged)     -> f32
        gemm_bf16t<0, false><<<gemm_blocks(N_CTX, D_MODEL), 256, 0, stream>>>(
            AVb, WOVt, nullptr, ATT, N_CTX, D_MODEL, D_MODEL, D_MODEL, D_MODEL, D_MODEL);
        // X += MLP + ATT
        add3_kernel<<<(int)(nXd / 256), 256, 0, stream>>>(X, MLP, ATT, (int)nXd);
    }

    // logits = X @ tok_emb^T
    convert_bf16_kernel<<<(int)(nXd / 1024), 256, 0, stream>>>(X, Xb, nXd);
    gemm_bf16t<0, false><<<gemm_blocks(N_CTX, D_VOCAB), 256, 0, stream>>>(
        Xb, TEb, nullptr, logits, N_CTX, D_VOCAB, D_MODEL, D_MODEL, D_MODEL, D_VOCAB);
}
